// GraphSAGE_local_6871947673826
// MI455X (gfx1250) — matrix operand order repaired and checked by reading, not compiled, not run on hardware
//
#include <hip/hip_runtime.h>
#include <hip/hip_bf16.h>
#include <stdint.h>

// ---------------------------------------------------------------------------
// GraphSAGE (2-layer) for MI455X / gfx1250, wave32, WMMA bf16 (f32 accum).
// - weights repacked once into the WMMA B-fragment layout (bf16, 32B/lane)
// - block's A rows staged into LDS via gfx1250 async copy
//   (global_load_async_to_lds_b128 + s_wait_asynccnt), read back as fragments
// - mean division folded into the f32 accumulator (not the A elements)
// - 2-tile-per-wave for M=256 passes: one A fragment -> two WMMAs
// ---------------------------------------------------------------------------

typedef __attribute__((ext_vector_type(16))) __bf16 v16bf;
typedef __attribute__((ext_vector_type(8)))  float  v8f;

// ---------------------------------------------------------------------------
// Zero-fill (agg buffers / counts must be re-initialized every call).
// ---------------------------------------------------------------------------
__global__ void zero_f32(float* __restrict__ p, long long n) {
  long long i   = (long long)blockIdx.x * blockDim.x + threadIdx.x;
  long long str = (long long)gridDim.x * blockDim.x;
  for (; i < n; i += str) p[i] = 0.0f;
}

// ---------------------------------------------------------------------------
// Repack W[K,M] fp32 row-major -> bf16 B-fragment layout:
//   P[ ((k/16)*M + m)*16 + (k%16) ] = bf16(W[k*M + m])
// so one lane's 16-K-deep fragment column is 32 contiguous bytes.
// ---------------------------------------------------------------------------
__global__ void pack_w_bf16(const float* __restrict__ W, __bf16* __restrict__ P,
                            int KM, int mshift) {
  const unsigned i = blockIdx.x * blockDim.x + threadIdx.x;
  if (i >= (unsigned)KM) return;
  const unsigned k = i >> mshift;
  const unsigned m = i & ((1u << mshift) - 1u);
  const unsigned M = 1u << mshift;
  P[(((k >> 4) * M + m) << 4) | (k & 15u)] = (__bf16)W[i];
}

// ---------------------------------------------------------------------------
// Build one A fragment (16 rows x 32 K, bf16) from an fp32 row slice (LDS).
// A layout (matches packed B): lanes 0-15 hold K 0-15, lanes 16-31 K 16-31.
// ---------------------------------------------------------------------------
__device__ __forceinline__ v16bf a_frag(const float* __restrict__ srow, int kh) {
  v16bf a;
#pragma unroll
  for (int j = 0; j < 8; ++j) {
    const int kk = (kh << 1) + (j << 1);
    const float2 v = *(const float2*)(srow + kk);     // ds_load_b64
    a[2 * j]     = (__bf16)v.x;
    a[2 * j + 1] = (__bf16)v.y;
  }
  return a;
}

// ---------------------------------------------------------------------------
// Fused SAGE GEMM (fully specialized):
//   OUT[N,M] = epilogue( (A1@W1)/deg + bias + A2@W2 ),  M = NT*128
//   epilogue = (optional L2-normalize per row) then ReLU
// Block = 256 threads = 8 waves, owns 16 rows x M cols; wave owns one
// (NT=1) or two (NT=2, cols n and n+128) 16x16 tiles sharing one A fragment.
// ---------------------------------------------------------------------------
template <int KDIM, int NT, bool HASCNT, bool HASA2, bool NORM>
__global__ __launch_bounds__(256)
void sage_gemm(const float* __restrict__ A1, const __bf16* __restrict__ W1p,
               const float* __restrict__ bias,
               const float* __restrict__ A2, const __bf16* __restrict__ W2p,
               const float* __restrict__ cnt,
               float* __restrict__ OUT)
{
  constexpr int M    = NT * 128;
  constexpr int SROW = KDIM + 4;           // padded LDS row stride (banks)
  constexpr int KSH  = (KDIM == 128) ? 7 : 8;
  constexpr int NMAT = HASA2 ? 2 : 1;

  __shared__ float sA[NMAT * 16 * SROW];
  __shared__ float s_sumsq[16];

  const int t    = threadIdx.x;
  const int wave = t >> 5;                 // 0..7
  const int lane = t & 31;
  const int l15  = lane & 15;
  const int hi   = lane >> 4;              // 0 | 1
  const int kh   = hi ? 8 : 0;
  const int row0 = blockIdx.x << 4;        // 16 rows per block

  // ---- stage block's A rows (contiguous 16*KDIM floats) into LDS, async ----
  {
    const unsigned base = (unsigned)row0 << KSH;   // row0*KDIM
    constexpr int ITER = (16 * KDIM) / (256 * 4);  // b128 per thread per iter
#pragma unroll
    for (int m = 0; m < NMAT; ++m) {
      const float* g = (m == 0 ? A1 : A2) + base;
      float*       l = sA + m * 16 * SROW;
#pragma unroll
      for (int it = 0; it < ITER; ++it) {
        const int idx = (t + it * 256) << 2;       // float index in 16xK tile
        const int row = idx >> KSH;
        const int col = idx & (KDIM - 1);
        const uint32_t loff = (uint32_t)(uintptr_t)(l + row * SROW + col);
        asm volatile("global_load_async_to_lds_b128 %0, %1, off"
                     :: "v"(loff), "v"(g + idx) : "memory");   // ASYNCcnt++
      }
    }
    asm volatile("s_wait_asynccnt 0x0" ::: "memory");
    __syncthreads();
  }

  const float* sR1 = sA + l15 * SROW;
  const float* sR2 = sA + 16 * SROW + l15 * SROW;  // only HASA2
  const unsigned ncol0 = (unsigned)((wave << 4) + l15);
  const unsigned ncol1 = ncol0 + 128u;             // NT==2 only

  v8f c0 = {0.f, 0.f, 0.f, 0.f, 0.f, 0.f, 0.f, 0.f};
  v8f c1 = {0.f, 0.f, 0.f, 0.f, 0.f, 0.f, 0.f, 0.f};

  // ---- GEMM 1: A1 @ W1 (raw sums; mean applied to accumulator below) ----
#pragma unroll
  for (int k0 = 0; k0 < KDIM; k0 += 32) {
    const v16bf a  = a_frag(sR1 + k0, kh);
    const unsigned kb = (unsigned)(k0 >> 4) + (unsigned)hi;
    const v16bf b0 = *(const v16bf*)(W1p + ((kb * M + ncol0) << 4));
    c0 = __builtin_amdgcn_wmma_f32_16x16x32_bf16(false, a, false, b0,
                                                 (short)0, c0, false, false);
    if (NT == 2) {
      const v16bf b1 = *(const v16bf*)(W1p + ((kb * M + ncol1) << 4));
      c1 = __builtin_amdgcn_wmma_f32_16x16x32_bf16(false, a, false, b1,
                                                   (short)0, c1, false, false);
    }
  }

  // ---- mean: (A1@W1)/deg — one scale per accumulator row ----
  if (HASCNT) {
#pragma unroll
    for (int q = 0; q < 8; ++q) {
      const float ic = 1.0f / fmaxf(cnt[row0 + q + kh], 1.0f);
      c0[q] *= ic;
      if (NT == 2) c1[q] *= ic;
    }
  }

  // ---- GEMM 2 (root path): + A2 @ W2, accumulated by WMMA C-operand ----
  if (HASA2) {
#pragma unroll
    for (int k0 = 0; k0 < KDIM; k0 += 32) {
      const v16bf a  = a_frag(sR2 + k0, kh);
      const unsigned kb = (unsigned)(k0 >> 4) + (unsigned)hi;
      const v16bf b0 = *(const v16bf*)(W2p + ((kb * M + ncol0) << 4));
      c0 = __builtin_amdgcn_wmma_f32_16x16x32_bf16(false, a, false, b0,
                                                   (short)0, c0, false, false);
      if (NT == 2) {
        const v16bf b1 = *(const v16bf*)(W2p + ((kb * M + ncol1) << 4));
        c1 = __builtin_amdgcn_wmma_f32_16x16x32_bf16(false, a, false, b1,
                                                     (short)0, c1, false, false);
      }
    }
  }

  // ---- epilogue: bias, (normalize), relu, store ----
  {
    const float bv0 = bias[ncol0];
#pragma unroll
    for (int q = 0; q < 8; ++q) c0[q] += bv0;
    if (NT == 2) {
      const float bv1 = bias[ncol1];
#pragma unroll
      for (int q = 0; q < 8; ++q) c1[q] += bv1;
    }
  }

  if (NORM) {                           // only instantiated with NT==1, M==128
    if (t < 16) s_sumsq[t] = 0.0f;
    __syncthreads();
#pragma unroll
    for (int q = 0; q < 8; ++q)
      atomicAdd(&s_sumsq[q + kh], c0[q] * c0[q]);    // ds_add_f32 reduce
    __syncthreads();
#pragma unroll
    for (int q = 0; q < 8; ++q) {
      const float nrm = sqrtf(s_sumsq[q + kh]);
      c0[q] *= 1.0f / fmaxf(nrm, 1e-12f);
    }
  }

#pragma unroll
  for (int q = 0; q < 8; ++q) {
    const unsigned row = (unsigned)(row0 + q + kh);  // C/D layout: VGPR q -> M=q(+8)
    OUT[row * M + ncol0] = fmaxf(c0[q], 0.0f);
    if (NT == 2)
      OUT[row * M + ncol1] = fmaxf(c1[q], 0.0f);
  }
}

// ---------------------------------------------------------------------------
// Edge scatter-sum, one wave per edge: endpoints loaded once per wave
// (uniform address -> single cacheline), each lane owns a float4 slice.
// f32 atomics land in the L2-resident agg buffer.
// ---------------------------------------------------------------------------
__global__ __launch_bounds__(256)
void sage_aggregate(const float* __restrict__ xp,
                    const long long* __restrict__ src,
                    const long long* __restrict__ dst,
                    float* __restrict__ agg,
                    float* __restrict__ cnt,   // nullptr to skip counting
                    int E, int D)
{
  const int lane = threadIdx.x & 31;
  const int e = blockIdx.x * (blockDim.x >> 5) + (threadIdx.x >> 5);
  if (e >= E) return;
  const unsigned s = (unsigned)src[e];
  const unsigned d = (unsigned)dst[e];
  const float* xs = xp + s * (unsigned)D;
  float*       ad = agg + d * (unsigned)D;
  for (int f = lane << 2; f < D; f += 128) {   // 1 iter (D=128) or 2 (D=256)
    const float4 v = *(const float4*)(xs + f);
    atomicAdd(ad + f + 0, v.x);
    atomicAdd(ad + f + 1, v.y);
    atomicAdd(ad + f + 2, v.z);
    atomicAdd(ad + f + 3, v.w);
  }
  if (cnt && lane == 0) atomicAdd(&cnt[d], 1.0f);
}

// ---------------------------------------------------------------------------
// Launch: pack weights -> layer1 (proj -> scatter -> fused lin_l/lin_r)
// -> layer2 with normalize. Counts computed once (same graph both layers).
// Workspace (fp32 region then packed-weight bf16 region):
//   regionA : N*256 f32  (xp1|agg1 in layer1, reused as xp2 in layer2)
//   cnt     : N     f32
//   h       : N*256 f32
//   agg2    : N*256 f32
//   packed  : 208K bf16 (six weight matrices, fragment layout)
// ---------------------------------------------------------------------------
extern "C" void kernel_launch(void* const* d_in, const int* in_sizes, int n_in,
                              void* d_out, int out_size, void* d_ws, size_t ws_size,
                              hipStream_t stream) {
  const float*      x   = (const float*)d_in[0];
  const long long*  ei  = (const long long*)d_in[1];
  const float*      w1p = (const float*)d_in[2];
  const float*      b1p = (const float*)d_in[3];
  const float*      w1l = (const float*)d_in[4];
  const float*      b1l = (const float*)d_in[5];
  const float*      w1r = (const float*)d_in[6];
  const float*      w2p = (const float*)d_in[7];
  const float*      b2p = (const float*)d_in[8];
  const float*      w2l = (const float*)d_in[9];
  const float*      b2l = (const float*)d_in[10];
  const float*      w2r = (const float*)d_in[11];

  const int N = in_sizes[0] / 128;
  const int E = in_sizes[1] / 2;
  const long long Nll = N;

  float* ws      = (float*)d_ws;
  float* regionA = ws;                       // N*256
  float* cntp    = regionA + Nll * 256;      // N
  float* h       = cntp + Nll;               // N*256
  float* agg2    = h + Nll * 256;            // N*256

  float* xp1  = regionA;                     // N*128
  float* agg1 = regionA + Nll * 128;         // N*128 (contiguous with cnt)
  float* xp2  = regionA;                     // N*256 (layer1 buffers dead)
  float* out  = (float*)d_out;

  // packed bf16 weights after the fp32 region (64B aligned)
  uintptr_t pkBase = (uintptr_t)(agg2 + Nll * 256);
  pkBase = (pkBase + 63) & ~(uintptr_t)63;
  __bf16* pk   = (__bf16*)pkBase;
  __bf16* p1p  = pk;                 // 128*128
  __bf16* p1l  = p1p + 128 * 128;    // 128*256
  __bf16* p1r  = p1l + 128 * 256;    // 128*256
  __bf16* p2p  = p1r + 128 * 256;    // 256*256
  __bf16* p2l  = p2p + 256 * 256;    // 256*128
  __bf16* p2r  = p2l + 256 * 128;    // 256*128

  // ---- one-time weight repack (tiny) ----
  pack_w_bf16<<<(128 * 128 + 255) / 256, 256, 0, stream>>>(w1p, p1p, 128 * 128, 7);
  pack_w_bf16<<<(128 * 256 + 255) / 256, 256, 0, stream>>>(w1l, p1l, 128 * 256, 8);
  pack_w_bf16<<<(128 * 256 + 255) / 256, 256, 0, stream>>>(w1r, p1r, 128 * 256, 8);
  pack_w_bf16<<<(256 * 256 + 255) / 256, 256, 0, stream>>>(w2p, p2p, 256 * 256, 8);
  pack_w_bf16<<<(256 * 128 + 255) / 256, 256, 0, stream>>>(w2l, p2l, 256 * 128, 7);
  pack_w_bf16<<<(256 * 128 + 255) / 256, 256, 0, stream>>>(w2r, p2r, 256 * 128, 7);

  // ---- layer 1 ----
  zero_f32<<<2048, 256, 0, stream>>>(agg1, Nll * 128 + Nll);  // agg1 + cnt

  // xp1 = relu(x @ w1_proj + b1_proj)          K=128, M=128
  sage_gemm<128, 1, false, false, false><<<N / 16, 256, 0, stream>>>(
      x, p1p, b1p, nullptr, nullptr, nullptr, xp1);

  sage_aggregate<<<(E + 7) / 8, 256, 0, stream>>>(xp1, ei, ei + E,
                                                  agg1, cntp, E, 128);

  // h = relu(mean @ w1_l + b1_l + x @ w1_r)    K=128, M=256
  sage_gemm<128, 2, true, true, false><<<N / 16, 256, 0, stream>>>(
      agg1, p1l, b1l, x, p1r, cntp, h);

  // ---- layer 2 (cnt reused: same graph) ----
  zero_f32<<<2048, 256, 0, stream>>>(agg2, Nll * 256);

  // xp2 = relu(h @ w2_proj + b2_proj)          K=256, M=256
  sage_gemm<256, 2, false, false, false><<<N / 16, 256, 0, stream>>>(
      h, p2p, b2p, nullptr, nullptr, nullptr, xp2);

  sage_aggregate<<<(E + 7) / 8, 256, 0, stream>>>(xp2, ei, ei + E,
                                                  agg2, nullptr, E, 256);

  // out = relu(normalize(mean @ w2_l + b2_l + h @ w2_r))   K=256, M=128
  sage_gemm<256, 1, true, true, true><<<N / 16, 256, 0, stream>>>(
      agg2, p2l, b2l, h, p2r, cntp, out);
}
